// GraphNeuralNetwork_19593640805088
// MI455X (gfx1250) — matrix operand order repaired and checked by reading, not compiled, not run on hardware
//
#include <hip/hip_runtime.h>

typedef __attribute__((ext_vector_type(16))) __bf16 v16bf;
typedef __attribute__((ext_vector_type(8)))  float  v8f;

// ---------------- degree / normalization ----------------
__global__ void k_deg_init(float* deg, int n) {
    int i = blockIdx.x * blockDim.x + threadIdx.x;
    if (i < n) deg[i] = 1.0f;                      // self loop contributes 1
}
__global__ void k_deg_count(const int* __restrict__ dst, float* deg, int e) {
    int i = blockIdx.x * blockDim.x + threadIdx.x;
    if (i < e) atomicAdd(&deg[dst[i]], 1.0f);
}
__global__ void k_dinv(float* deg, int n) {
    int i = blockIdx.x * blockDim.x + threadIdx.x;
    if (i < n) { float d = deg[i]; deg[i] = d > 0.f ? rsqrtf(d) : 0.f; }
}

// -------- pack fp32 W (K x dout, row-major) into per-lane WMMA-B bf16 frags --------
// layout: [ktile*ntiles + ntile][lane 0..31][i 0..15]
//   lane: n = ntile*16 + (lane&15), k = ktile*32 + (i>>3)*16 + (lane>>4)*8 + (i&7)
__global__ void k_pack_w(const float* __restrict__ W, __bf16* __restrict__ Wp,
                         int K, int dout) {
    int idx = blockIdx.x * blockDim.x + threadIdx.x;
    if (idx >= K * dout) return;
    int i      = idx & 15;
    int lane   = (idx >> 4) & 31;
    int tile   = idx >> 9;
    int ntiles = dout >> 4;
    int ntile  = tile % ntiles;
    int ktile  = tile / ntiles;
    int n = ntile * 16 + (lane & 15);
    int k = ktile * 32 + (i >> 3) * 16 + (lane >> 4) * 8 + (i & 7);
    Wp[idx] = (__bf16)W[k * dout + n];
}

// ---------------- H = X @ W via v_wmma_f32_16x16x32_bf16 ----------------
// grid: (ceil(N/128), dout/16), block: 256 (8 waves, wave w owns rows [128bx+16w, +16))
__global__ void __launch_bounds__(256) k_gemm_wmma(
    const float* __restrict__ X, const __bf16* __restrict__ Wp,
    float* __restrict__ H, int N, int K, int dout) {
    int lane = threadIdx.x & 31;
    int wv   = threadIdx.x >> 5;
    int m    = lane & 15;
    int half = lane >> 4;
    int by   = blockIdx.y;
    int ntiles = dout >> 4;
    long row = (long)blockIdx.x * 128 + wv * 16 + m;
    const float* Xrow = X + row * (long)K;
    v8f c = {};
    for (int kk = 0; kk < K; kk += 32) {
        v16bf a;
        if (row < N) {
            const float* p0 = Xrow + kk + half * 8;        // K = kk + half*8 .. +7
            const float* p1 = Xrow + kk + 16 + half * 8;   // K = kk+16+half*8 .. +7
#pragma unroll
            for (int e = 0; e < 8; ++e) a[e]     = (__bf16)p0[e];
#pragma unroll
            for (int e = 0; e < 8; ++e) a[8 + e] = (__bf16)p1[e];
        } else {
#pragma unroll
            for (int e = 0; e < 16; ++e) a[e] = (__bf16)0.f;
        }
        v16bf b = *(const v16bf*)(Wp + (((size_t)(kk >> 5) * ntiles + by) * 32 + lane) * 16);
        c = __builtin_amdgcn_wmma_f32_16x16x32_bf16(
                false, a, false, b, (short)0, c, false, false);
    }
    int col = by * 16 + (lane & 15);
    long rbase = (long)blockIdx.x * 128 + wv * 16 + half * 8;
#pragma unroll
    for (int r = 0; r < 8; ++r) {
        long rr = rbase + r;
        if (rr < N) H[rr * dout + col] = c[r];
    }
}

// ---------------- agg = h * dinv^2 (self loop) ----------------
__global__ void k_self_init(const float* __restrict__ H, const float* __restrict__ dinv,
                            float* __restrict__ agg, long tot, int dq) {
    long tid = (long)blockIdx.x * blockDim.x + threadIdx.x;
    if (tid >= tot) return;
    long i = tid / dq; int cidx = (int)(tid % dq);
    float s = dinv[i]; s *= s;
    float4 h = ((const float4*)H)[i * dq + cidx];
    float4 o; o.x = h.x * s; o.y = h.y * s; o.z = h.z * s; o.w = h.w * s;
    ((float4*)agg)[i * dq + cidx] = o;
}

// ---------------- agg[dst] += h[src] * dinv[src]*dinv[dst] ----------------
__global__ void k_scatter(const float* __restrict__ H, const int* __restrict__ src,
                          const int* __restrict__ dst, const float* __restrict__ dinv,
                          float* __restrict__ agg, long tot, int d, int dq) {
    long tid = (long)blockIdx.x * blockDim.x + threadIdx.x;
    if (tid >= tot) return;
    long ei = tid / dq; int cidx = (int)(tid % dq);
    int s = src[ei], t = dst[ei];
    float nrm = dinv[s] * dinv[t];
    float4 h = ((const float4*)(H + (size_t)s * d))[cidx];
    float* ap = agg + (size_t)t * d + (size_t)cidx * 4;
    atomicAdd(ap + 0, h.x * nrm);
    atomicAdd(ap + 1, h.y * nrm);
    atomicAdd(ap + 2, h.z * nrm);
    atomicAdd(ap + 3, h.w * nrm);
}

// ---------------- out = (agg + b) [+ relu] ----------------
__global__ void k_bias_act(const float* __restrict__ agg, const float* __restrict__ b,
                           float* __restrict__ out, long tot, int dq, int relu) {
    long tid = (long)blockIdx.x * blockDim.x + threadIdx.x;
    if (tid >= tot) return;
    long i = tid / dq; int cidx = (int)(tid % dq);
    float4 v = ((const float4*)agg)[i * dq + cidx];
    v.x += b[cidx * 4 + 0]; v.y += b[cidx * 4 + 1];
    v.z += b[cidx * 4 + 2]; v.w += b[cidx * 4 + 3];
    if (relu) {
        v.x = fmaxf(v.x, 0.f); v.y = fmaxf(v.y, 0.f);
        v.z = fmaxf(v.z, 0.f); v.w = fmaxf(v.w, 0.f);
    }
    ((float4*)out)[i * dq + cidx] = v;
}

__global__ void k_zero(float* p, int n) {
    int i = blockIdx.x * blockDim.x + threadIdx.x;
    if (i < n) p[i] = 0.f;
}

// ---------------- global mean pool accumulation ----------------
__global__ void k_pool(const float* __restrict__ h, const int* __restrict__ batch,
                       float* __restrict__ sum, float* __restrict__ cnt, long tot) {
    long tid = (long)blockIdx.x * blockDim.x + threadIdx.x;
    if (tid >= tot) return;
    long i = tid >> 5; int j = (int)(tid & 31);
    int g = batch[i];
    atomicAdd(&sum[g * 32 + j], h[i * 32 + j]);
    if (j == 0) atomicAdd(&cnt[g], 1.0f);
}

// ---------------- tiny MLP head: 32 -> 64 -> 16 -> 8, one thread per graph ----------------
__global__ void k_mlp(const float* __restrict__ sum, const float* __restrict__ cnt,
                      const float* __restrict__ M1w, const float* __restrict__ M1b,
                      const float* __restrict__ M2w, const float* __restrict__ M2b,
                      const float* __restrict__ M3w, const float* __restrict__ M3b,
                      float* __restrict__ out) {
    int g = threadIdx.x;
    if (g >= 64) return;
    float inv = 1.0f / fmaxf(cnt[g], 1.0f);
    float gv[32];
#pragma unroll
    for (int j = 0; j < 32; ++j) gv[j] = sum[g * 32 + j] * inv;
    float h1[64];
    for (int o = 0; o < 64; ++o) {
        float acc = M1b[o];
        for (int k = 0; k < 32; ++k) acc += gv[k] * M1w[k * 64 + o];
        h1[o] = fmaxf(acc, 0.f);
    }
    float h2[16];
    for (int o = 0; o < 16; ++o) {
        float acc = M2b[o];
        for (int k = 0; k < 64; ++k) acc += h1[k] * M2w[k * 16 + o];
        h2[o] = fmaxf(acc, 0.f);
    }
    for (int o = 0; o < 8; ++o) {
        float acc = M3b[o];
        for (int k = 0; k < 16; ++k) acc += h2[k] * M3w[k * 8 + o];
        out[g * 8 + o] = acc;
    }
}

extern "C" void kernel_launch(void* const* d_in, const int* in_sizes, int n_in,
                              void* d_out, int out_size, void* d_ws, size_t ws_size,
                              hipStream_t stream) {
    const float* x    = (const float*)d_in[0];
    const int*   ei   = (const int*)d_in[1];
    const int*   bat  = (const int*)d_in[2];
    const float* W1 = (const float*)d_in[3];  const float* b1 = (const float*)d_in[4];
    const float* W2 = (const float*)d_in[5];  const float* b2 = (const float*)d_in[6];
    const float* W3 = (const float*)d_in[7];  const float* b3 = (const float*)d_in[8];
    const float* W4 = (const float*)d_in[9];  const float* b4 = (const float*)d_in[10];
    const float* Wo = (const float*)d_in[11]; const float* bo = (const float*)d_in[12];
    const float* M1w = (const float*)d_in[13]; const float* M1b = (const float*)d_in[14];
    const float* M2w = (const float*)d_in[15]; const float* M2b = (const float*)d_in[16];
    const float* M3w = (const float*)d_in[17]; const float* M3b = (const float*)d_in[18];

    const int N = in_sizes[2];
    const int E = in_sizes[1] / 2;
    const int IN = in_sizes[0] / N;        // 64
    const int HID = in_sizes[4];           // 128
    const int OUT = in_sizes[12];          // 32

    // workspace layout (floats), Nr keeps every region 512B-aligned
    long Nr = ((long)N + 127) & ~127L;
    float* ws   = (float*)d_ws;
    float* dinv = ws;
    float* bufX = dinv + Nr;               // current features (N x <=128)
    float* bufH = bufX + Nr * 128;         // GEMM output
    float* bufA = bufH + Nr * 128;         // aggregation
    float* psum = bufA + Nr * 128;         // 64 x 32
    float* pcnt = psum + 64 * 32;          // 64
    __bf16* Wp  = (__bf16*)(pcnt + 128);   // packed weights (<= 128*128 bf16)

    const int* src = ei;
    const int* dst = ei + E;
    const int T = 256;

    // symmetric degree normalization D^{-1/2}(A+I)D^{-1/2}
    k_deg_init<<<(N + T - 1) / T, T, 0, stream>>>(dinv, N);
    k_deg_count<<<(E + T - 1) / T, T, 0, stream>>>(dst, dinv, E);
    k_dinv<<<(N + T - 1) / T, T, 0, stream>>>(dinv, N);

    auto layer = [&](const float* Xin, int K, const float* W, const float* b,
                     int dout, int relu, float* Xout) {
        k_pack_w<<<(K * dout + T - 1) / T, T, 0, stream>>>(W, Wp, K, dout);
        dim3 grid((N + 127) / 128, dout / 16);
        k_gemm_wmma<<<grid, 256, 0, stream>>>(Xin, Wp, bufH, N, K, dout);
        int dq = dout / 4;
        long ntot = (long)N * dq;
        long etot = (long)E * dq;
        k_self_init<<<(ntot + T - 1) / T, T, 0, stream>>>(bufH, dinv, bufA, ntot, dq);
        k_scatter<<<(etot + T - 1) / T, T, 0, stream>>>(bufH, src, dst, dinv, bufA, etot, dout, dq);
        k_bias_act<<<(ntot + T - 1) / T, T, 0, stream>>>(bufA, b, Xout, ntot, dq, relu);
    };

    layer(x,    IN,  W1, b1, HID, 1, bufX);
    layer(bufX, HID, W2, b2, HID, 1, bufX);
    layer(bufX, HID, W3, b3, HID, 1, bufX);
    layer(bufX, HID, W4, b4, HID, 1, bufX);
    layer(bufX, HID, Wo, bo, OUT, 0, bufX);   // final: N x 32, no relu

    // global mean pool + MLP head
    k_zero<<<(64 * 32 + 64 + T - 1) / T, T, 0, stream>>>(psum, 64 * 32 + 64);
    long ptot = (long)N * 32;
    k_pool<<<(ptot + T - 1) / T, T, 0, stream>>>(bufX, bat, psum, pcnt, ptot);
    k_mlp<<<1, 64, 0, stream>>>(psum, pcnt, M1w, M1b, M2w, M2b, M3w, M3b, (float*)d_out);
}
